// Plan2VecEncoder_67929202753940
// MI455X (gfx1250) — compile-verified, matrix operand order checked by reading, not run on hardware
//
#include <hip/hip_runtime.h>
#include <hip/hip_bf16.h>

// ---------------- problem constants ----------------
#define NN      200000
#define EE      800000
#define GG      2048
#define SS      128
#define HH      256
#define TEXTD   64
#define OUTD    512
#define XCOLS   13          // 1 + NUMERIC
#define CATW    352         // 323 padded to multiple of 32
#define CATREAL 323

typedef __bf16 bf16_t;
typedef __attribute__((ext_vector_type(16))) __bf16 v16bf;
typedef __attribute__((ext_vector_type(8)))  float  v8f;

// gfx1250 async global->LDS copy path (ASYNCcnt-tracked), with sync fallback
#if defined(__has_builtin)
#if __has_builtin(__builtin_amdgcn_global_load_async_to_lds_b64)
#define ASYNC_LDS 1
#endif
#endif

typedef int v2i __attribute__((vector_size(8)));
typedef __attribute__((address_space(1))) v2i* as1_v2i;
typedef __attribute__((address_space(3))) v2i* as3_v2i;

__device__ __forceinline__ void asyncWait0() {
#if defined(__has_builtin) && __has_builtin(__builtin_amdgcn_s_wait_asynccnt)
    __builtin_amdgcn_s_wait_asynccnt(0);
#else
    asm volatile("s_wait_asynccnt 0x0" ::: "memory");
#endif
}

__device__ __forceinline__ bf16_t f2bf(float f) {
    unsigned u = __builtin_bit_cast(unsigned, f);
    unsigned r = u + 0x7FFFu + ((u >> 16) & 1u);   // round-to-nearest-even
    unsigned short s = (unsigned short)(r >> 16);
    return __builtin_bit_cast(bf16_t, s);
}

__device__ __forceinline__ void atomAddF(float* p, float v) {
    __hip_atomic_fetch_add(p, v, __ATOMIC_RELAXED, __HIP_MEMORY_SCOPE_AGENT);
}

// ---------------- small utility kernels ----------------
__global__ void zero_f32(float* p, long long n) {
    long long g = (long long)blockIdx.x * blockDim.x + threadIdx.x;
    if (g < n) p[g] = 0.f;
}

// fp32 -> bf16 weight copy with zero-padded K (columns)
__global__ void wconv_kernel(const float* __restrict__ src, bf16_t* __restrict__ dst,
                             int rows, int scols, int dcols) {
    long long g = (long long)blockIdx.x * blockDim.x + threadIdx.x;
    long long n = (long long)rows * dcols;
    if (g >= n) return;
    int r = (int)(g / dcols);
    int c = (int)(g - (long long)r * dcols);
    dst[g] = f2bf(c < scols ? src[(size_t)r * scols + c] : 0.f);
}

// bias[t][j] = sum_k edge_embed[t][k]*lw[j][k] + lb[j]   (t<4, j<d, stored stride 256)
__global__ void __launch_bounds__(256) etype_bias_kernel(const float* __restrict__ eemb,
                                                         const float* __restrict__ lw,
                                                         const float* __restrict__ lb,
                                                         float* __restrict__ bias, int d) {
    int t = blockIdx.x, j = threadIdx.x;
    float acc = 0.f;
    if (j < d) {
        acc = lb[j];
        #pragma unroll
        for (int k = 0; k < 16; ++k) acc += eemb[t * 16 + k] * lw[j * 16 + k];
    }
    bias[t * HH + j] = acc;
}

// h0 = [op_embed[op_id] (32) | numerics (12) | zeros] per node, stride 256
__global__ void init_h_kernel(const float* __restrict__ x, const float* __restrict__ opemb,
                              float* __restrict__ h) {
    long long g = (long long)blockIdx.x * blockDim.x + threadIdx.x;
    if (g >= (long long)NN * HH) return;
    int i = (int)(g >> 8), j = (int)(g & 255);
    float v = 0.f;
    if (j < 32) {
        int op = (int)x[(size_t)i * XCOLS];
        op = op < 0 ? 0 : (op > 63 ? 63 : op);
        v = opemb[op * 32 + j];
    } else if (j < 44) {
        v = x[(size_t)i * XCOLS + 1 + (j - 32)];
    }
    h[g] = v;
}

// aggr[dst] += relu(h[src] + bias[etype])
__global__ void scatter_kernel(const float* __restrict__ h, const float* __restrict__ bias,
                               const int* __restrict__ src, const int* __restrict__ dst,
                               const int* __restrict__ eattr, float* __restrict__ aggr,
                               long long total, int d) {
    long long g = (long long)blockIdx.x * blockDim.x + threadIdx.x;
    if (g >= total) return;
    int e, j;
    if (d == 256) { e = (int)(g >> 8); j = (int)(g & 255); }
    else          { e = (int)(g / d);  j = (int)(g - (long long)e * d); }
    int t = eattr[e]; t = t > 3 ? 3 : t;
    float m = h[(size_t)src[e] * HH + j] + bias[t * HH + j];
    if (m > 0.f) atomAddF(&aggr[(size_t)dst[e] * HH + j], m);
}

// z16 = bf16((1+eps)*h + aggr)
__global__ void z_convert_kernel(const float* __restrict__ h, const float* __restrict__ aggr,
                                 const float* __restrict__ eps_p, bf16_t* __restrict__ z16,
                                 long long n) {
    long long g = (long long)blockIdx.x * blockDim.x + threadIdx.x;
    if (g >= n) return;
    float e = 1.f + eps_p[0];
    z16[g] = f2bf(e * h[g] + aggr[g]);
}

// ------------- WMMA bf16 GEMM: C(M x N) = act(A(M x K) * B(N x K)^T + bias) -------------
// block = 128 threads (4 waves). block computes a 16-row x 256-col stripe.
// A tile double-buffered in LDS via async global->LDS copies (ASYNCcnt), 1 barrier/K-step.
// grid.x = M/16, grid.y = N/256.
#define MODE_NONE 0
#define MODE_RELU 1
#define MODE_LEAKY 2
__global__ void __launch_bounds__(128)
gemm_bf16_wmma(const bf16_t* __restrict__ A, int lda,
               const bf16_t* __restrict__ B, int ldb,
               const float* __restrict__ bias,
               float* __restrict__ outF, bf16_t* __restrict__ outB,
               int ldc, int K, int mode) {
    __shared__ bf16_t As[2][16 * 32];
    int tid  = threadIdx.x;
    int wave = tid >> 5;
    int lane = tid & 31;
    int m0   = blockIdx.x * 16;
    int nblk = blockIdx.y * 256 + wave * 64;
    int kb   = (lane < 16) ? 0 : 8;   // CDNA5 wave32 16-bit A/B fragment K-offset
    int mrow = lane & 15;

    // per-thread slice of the 16x32 A tile: 4 bf16 = 8 bytes
    int r    = tid >> 3;
    int cidx = (tid & 7) << 2;
    const bf16_t* gA = A + (size_t)(m0 + r) * lda + cidx;

    v8f c[4];
    #pragma unroll
    for (int t = 0; t < 4; ++t) c[t] = (v8f){0.f,0.f,0.f,0.f,0.f,0.f,0.f,0.f};

#ifdef ASYNC_LDS
    auto issue = [&](int nbuf, int k0) {
        __builtin_amdgcn_global_load_async_to_lds_b64(
            (as1_v2i)(void*)(gA + k0),
            (as3_v2i)(void*)&As[nbuf][r * 32 + cidx], 0, 0);
    };
    issue(0, 0);
#endif

    int buf = 0;
    for (int k0 = 0; k0 < K; k0 += 32) {
#ifdef ASYNC_LDS
        asyncWait0();          // our async writes into As[buf] have landed
        __syncthreads();       // everyone's have; prev compute done -> safe to overwrite other buf
        if (k0 + 32 < K) issue(buf ^ 1, k0 + 32);  // prefetch next tile under the WMMAs
#else
        {
            const bf16_t* s = gA + k0;
            As[buf][r * 32 + cidx + 0] = s[0];
            As[buf][r * 32 + cidx + 1] = s[1];
            As[buf][r * 32 + cidx + 2] = s[2];
            As[buf][r * 32 + cidx + 3] = s[3];
        }
        __syncthreads();
#endif

        v16bf a;
        #pragma unroll
        for (int i = 0; i < 8; ++i) {
            a[i]     = As[buf][mrow * 32 + kb + i];
            a[i + 8] = As[buf][mrow * 32 + 16 + kb + i];
        }

        #pragma unroll
        for (int t = 0; t < 4; ++t) {
            int n = nblk + t * 16 + (lane & 15);
            const bf16_t* bp = B + (size_t)n * ldb + k0;
            v16bf b;
            #pragma unroll
            for (int i = 0; i < 8; ++i) {
                b[i]     = bp[kb + i];
                b[i + 8] = bp[16 + kb + i];
            }
            c[t] = __builtin_amdgcn_wmma_f32_16x16x32_bf16(
                       false, a, false, b, (short)0, c[t], false, false);
        }
#ifndef ASYNC_LDS
        __syncthreads();
#endif
        buf ^= 1;
    }

    // epilogue: C 16x16 fp32 layout -> VGPR j: row = j (+8 for hi lanes), col = lane&15
    #pragma unroll
    for (int t = 0; t < 4; ++t) {
        int col = nblk + t * 16 + (lane & 15);
        float bv = bias[col];
        #pragma unroll
        for (int j = 0; j < 8; ++j) {
            int rr = m0 + ((lane < 16) ? j : j + 8);
            float v = c[t][j] + bv;
            if (mode == MODE_RELU)       v = v > 0.f ? v : 0.f;
            else if (mode == MODE_LEAKY) v = v > 0.f ? v : 0.1f * v;
            if (outB) outB[(size_t)rr * ldc + col] = f2bf(v);
            else      outF[(size_t)rr * ldc + col] = v;
        }
    }
}

// LayerNorm (+ optional residual) + leaky, H=256, one row per 256-thread block
__global__ void __launch_bounds__(256)
ln_res_kernel(const float* __restrict__ hnew, float* __restrict__ h, int residual) {
    __shared__ float red[256];
    int row = blockIdx.x, tid = threadIdx.x;
    float v = hnew[(size_t)row * HH + tid];
    red[tid] = v;  __syncthreads();
    for (int s = 128; s > 0; s >>= 1) { if (tid < s) red[tid] += red[tid + s]; __syncthreads(); }
    float mean = red[0] * (1.f / HH);
    __syncthreads();
    float d = v - mean;
    red[tid] = d * d;  __syncthreads();
    for (int s = 128; s > 0; s >>= 1) { if (tid < s) red[tid] += red[tid + s]; __syncthreads(); }
    float var = red[0] * (1.f / HH);
    float o = d * rsqrtf(var + 1e-5f);
    if (residual) o += h[(size_t)row * HH + tid];
    o = o > 0.f ? o : 0.1f * o;
    h[(size_t)row * HH + tid] = o;
}

__global__ void pool_kernel(const float* __restrict__ h, const int* __restrict__ batch,
                            float* __restrict__ gpool) {
    long long g = (long long)blockIdx.x * blockDim.x + threadIdx.x;
    if (g >= (long long)NN * HH) return;
    int i = (int)(g >> 8), j = (int)(g & 255);
    atomAddF(&gpool[(size_t)batch[i] * HH + j], h[g]);
}

__global__ void stats_kernel(const float* __restrict__ x, const int* __restrict__ batch,
                             float* __restrict__ st) {   // st: [count | fan | cost] x G
    int i = blockIdx.x * blockDim.x + threadIdx.x;
    if (i >= NN) return;
    int b = batch[i];
    atomAddF(&st[b], 1.f);
    atomAddF(&st[GG + b],     x[(size_t)i * XCOLS + 5]);
    atomAddF(&st[2 * GG + b], x[(size_t)i * XCOLS + 4]);
}

// masked-mean text feature, written straight into cat16 cols [259,323)
__global__ void __launch_bounds__(64)
text_kernel(const int* __restrict__ ids, const float* __restrict__ mask,
            const float* __restrict__ tok, bf16_t* __restrict__ cat16) {
    int g = blockIdx.x, j = threadIdx.x;
    float acc = 0.f, len = 0.f;
    for (int s = 0; s < SS; ++s) {
        float m = mask[(size_t)g * SS + s];
        len += m;
        if (m != 0.f) {
            int id = ids[(size_t)g * SS + s];
            acc += tok[(size_t)id * TEXTD + j] * m;
        }
    }
    float L = len > 1.f ? len : 1.f;
    cat16[(size_t)g * CATW + 259 + j] = f2bf(acc / L);
}

// cat16 cols [0,259) and padding [323,352)
__global__ void cat_kernel(const float* __restrict__ gpool, const float* __restrict__ st,
                           bf16_t* __restrict__ cat16) {
    long long gid = (long long)blockIdx.x * blockDim.x + threadIdx.x;
    if (gid >= (long long)GG * CATW) return;
    int g = (int)(gid / CATW);
    int j = (int)(gid - (long long)g * CATW);
    if (j >= 259 && j < 259 + TEXTD) return;   // text_kernel owns these
    float v;
    if (j < 256)        v = gpool[(size_t)g * HH + j];
    else if (j == 256)  v = st[g];
    else if (j == 257)  { float c = st[g]; c = c > 1.f ? c : 1.f; v = st[GG + g] / c; }
    else if (j == 258)  { float c = st[g]; c = c > 1.f ? c : 1.f; v = st[2 * GG + g] / c; }
    else                v = 0.f;               // K padding
    cat16[gid] = f2bf(v);
}

// ---------------- host-side orchestration ----------------
extern "C" void kernel_launch(void* const* d_in, const int* in_sizes, int n_in,
                              void* d_out, int out_size, void* d_ws, size_t ws_size,
                              hipStream_t stream) {
    const float* x      = (const float*)d_in[0];
    const int*   eidx   = (const int*)  d_in[1];   // [2,E]
    const int*   eattr  = (const int*)  d_in[2];
    const int*   batch  = (const int*)  d_in[3];
    const int*   sqlids = (const int*)  d_in[4];
    const float* sqlmsk = (const float*)d_in[5];
    const float* opemb  = (const float*)d_in[6];
    const float* eemb   = (const float*)d_in[7];
    const float* tokemb = (const float*)d_in[8];
    const float* lw[3]  = {(const float*)d_in[9],  (const float*)d_in[16], (const float*)d_in[23]};
    const float* lb[3]  = {(const float*)d_in[10], (const float*)d_in[17], (const float*)d_in[24]};
    const float* eps[3] = {(const float*)d_in[11], (const float*)d_in[18], (const float*)d_in[25]};
    const float* W1[3]  = {(const float*)d_in[12], (const float*)d_in[19], (const float*)d_in[26]};
    const float* b1[3]  = {(const float*)d_in[13], (const float*)d_in[20], (const float*)d_in[27]};
    const float* W2[3]  = {(const float*)d_in[14], (const float*)d_in[21], (const float*)d_in[28]};
    const float* b2[3]  = {(const float*)d_in[15], (const float*)d_in[22], (const float*)d_in[29]};
    const float* mW1 = (const float*)d_in[30];
    const float* mb1 = (const float*)d_in[31];
    const float* mW2 = (const float*)d_in[32];
    const float* mb2 = (const float*)d_in[33];

    // workspace carve-up
    char* wsb = (char*)d_ws;
    size_t off = 0;
    auto carve = [&](size_t bytes) -> void* {
        off = (off + 255) & ~(size_t)255;
        void* p = wsb + off;
        off += bytes;
        return p;
    };
    float*  h     = (float*) carve((size_t)NN * HH * 4);
    float*  aggr  = (float*) carve((size_t)NN * HH * 4);
    bf16_t* z16   = (bf16_t*)carve((size_t)NN * HH * 2);
    bf16_t* t16   = (bf16_t*)carve((size_t)NN * HH * 2);
    const int Kl[3]  = {64, 256, 256};   // padded K per layer
    const int din[3] = {44, 256, 256};   // true input dim per layer
    bf16_t* w1b[3]; bf16_t* w2b[3]; float* ebias[3];
    for (int l = 0; l < 3; ++l) {
        w1b[l]   = (bf16_t*)carve((size_t)HH * Kl[l] * 2);
        w2b[l]   = (bf16_t*)carve((size_t)HH * HH * 2);
        ebias[l] = (float*) carve((size_t)4 * HH * 4);
    }
    bf16_t* mW1b  = (bf16_t*)carve((size_t)HH * CATW * 2);
    bf16_t* mW2b  = (bf16_t*)carve((size_t)OUTD * HH * 2);
    float*  gpool = (float*) carve((size_t)GG * HH * 4);
    float*  st    = (float*) carve((size_t)3 * GG * 4);
    bf16_t* cat16 = (bf16_t*)carve((size_t)GG * CATW * 2);
    bf16_t* hid16 = (bf16_t*)carve((size_t)GG * HH * 2);

    const int TB = 256;
    auto blocks = [](long long n, int tb) { return (unsigned)((n + tb - 1) / tb); };

    // weight conversion (fp32 -> bf16, zero K-padding) + per-edge-type bias tables
    for (int l = 0; l < 3; ++l) {
        wconv_kernel<<<blocks((long long)HH * Kl[l], TB), TB, 0, stream>>>(W1[l], w1b[l], HH, din[l], Kl[l]);
        wconv_kernel<<<blocks((long long)HH * HH,    TB), TB, 0, stream>>>(W2[l], w2b[l], HH, HH, HH);
        etype_bias_kernel<<<4, 256, 0, stream>>>(eemb, lw[l], lb[l], ebias[l], din[l]);
    }
    wconv_kernel<<<blocks((long long)HH * CATW, TB), TB, 0, stream>>>(mW1, mW1b, HH, CATREAL, CATW);
    wconv_kernel<<<blocks((long long)OUTD * HH, TB), TB, 0, stream>>>(mW2, mW2b, OUTD, HH, HH);

    // node features
    init_h_kernel<<<blocks((long long)NN * HH, TB), TB, 0, stream>>>(x, opemb, h);

    const int* esrc = eidx;
    const int* edst = eidx + EE;

    // 3 conv layers
    for (int l = 0; l < 3; ++l) {
        zero_f32<<<blocks((long long)NN * HH, TB), TB, 0, stream>>>(aggr, (long long)NN * HH);
        long long sc = (long long)EE * din[l];
        scatter_kernel<<<blocks(sc, TB), TB, 0, stream>>>(h, ebias[l], esrc, edst, eattr, aggr, sc, din[l]);
        z_convert_kernel<<<blocks((long long)NN * HH, TB), TB, 0, stream>>>(h, aggr, eps[l], z16, (long long)NN * HH);
        // t = relu(z @ W1^T + b1)  -> bf16
        gemm_bf16_wmma<<<dim3(NN / 16, 1), 128, 0, stream>>>(z16, HH, w1b[l], Kl[l], b1[l],
                                                             nullptr, t16, HH, Kl[l], MODE_RELU);
        // hnew = t @ W2^T + b2     -> fp32 (into aggr)
        gemm_bf16_wmma<<<dim3(NN / 16, 1), 128, 0, stream>>>(t16, HH, w2b[l], HH, b2[l],
                                                             aggr, nullptr, HH, HH, MODE_NONE);
        // h = leaky(LN(hnew) [+ h])
        ln_res_kernel<<<NN, 256, 0, stream>>>(aggr, h, l > 0 ? 1 : 0);
    }

    // pooling + stats + text + concat
    zero_f32<<<blocks((long long)GG * HH, TB), TB, 0, stream>>>(gpool, (long long)GG * HH);
    zero_f32<<<blocks((long long)3 * GG,  TB), TB, 0, stream>>>(st,    (long long)3 * GG);
    pool_kernel<<<blocks((long long)NN * HH, TB), TB, 0, stream>>>(h, batch, gpool);
    stats_kernel<<<blocks(NN, TB), TB, 0, stream>>>(x, batch, st);
    text_kernel<<<GG, 64, 0, stream>>>(sqlids, sqlmsk, tokemb, cat16);
    cat_kernel<<<blocks((long long)GG * CATW, TB), TB, 0, stream>>>(gpool, st, cat16);

    // head: hid = leaky(cat @ mW1^T + mb1) -> bf16 ; out = hid @ mW2^T + mb2 -> fp32
    gemm_bf16_wmma<<<dim3(GG / 16, 1), 128, 0, stream>>>(cat16, CATW, mW1b, CATW, mb1,
                                                         nullptr, hid16, HH, CATW, MODE_LEAKY);
    gemm_bf16_wmma<<<dim3(GG / 16, OUTD / 256), 128, 0, stream>>>(hid16, HH, mW2b, HH, mb2,
                                                                  (float*)d_out, nullptr, OUTD, HH, MODE_NONE);
}
